// Module_7954279432702
// MI455X (gfx1250) — matrix operand order repaired and checked by reading, not compiled, not run on hardware
//
#include <hip/hip_runtime.h>

// ---------------------------------------------------------------------------
// MoE-LoRA fused layer for MI455X (gfx1250, wave32, WMMA).
//   out = x @ W^T + b + sum_e gate_e * B_e @ (A_e @ x)
// Round 3: W staging in kernel 3 now uses the CDNA5 async VMEM->LDS path
// (global_load_async_to_lds_b128 + s_wait_asynccnt) instead of the
// VGPR round-trip; everything else as round 2 (pre-packed f16 operands,
// pure b128 -> v_wmma inner loops).
// ---------------------------------------------------------------------------

typedef _Float16 v16h __attribute__((ext_vector_type(16)));
typedef _Float16 h8   __attribute__((ext_vector_type(8)));
typedef _Float16 h4   __attribute__((ext_vector_type(4)));
typedef float    v8f  __attribute__((ext_vector_type(8)));
typedef float    f4   __attribute__((ext_vector_type(4)));

#define M_TOT 16384   // B*T = 8*2048
#define DDIM  1024
#define NEXP  8
#define RANK  16
#define ERDIM 128     // NEXP*RANK

// ---------------------------------------------------------------------------
// CDNA5 async VMEM->LDS helpers
// ---------------------------------------------------------------------------
__device__ __forceinline__ unsigned lds_off32(const void* p) {
  // LDS aperture: flat ADDR[31:0] is the LDS byte offset
  return (unsigned)(uintptr_t)p;
}

__device__ __forceinline__ void async_fill_b128(unsigned lds_off,
                                                const void* gaddr) {
  asm volatile("global_load_async_to_lds_b128 %0, %1, off"
               :
               : "v"(lds_off), "v"((unsigned long long)(uintptr_t)gaddr)
               : "memory");
}

__device__ __forceinline__ void wait_async0() {
  asm volatile("s_wait_asynccnt 0x0" ::: "memory");
}

// ---------------------------------------------------------------------------
// f16 fragment builders (operands pre-packed, contiguous, 16B aligned)
// ---------------------------------------------------------------------------
__device__ __forceinline__ v16h combine16(h8 lo, h8 hi) {
  return __builtin_shufflevector(lo, hi, 0, 1, 2, 3, 4, 5, 6, 7,
                                 8, 9, 10, 11, 12, 13, 14, 15);
}

// A-operand (16x32): lane m=lane&15, koff=16*(lane>>4);
// elements 0..15 = K koff+0..15.
__device__ __forceinline__ v16h frag_a_h(const _Float16* __restrict__ base,
                                         int ld, int lane) {
  const int m    = lane & 15;
  const int koff = (lane >> 4) << 4;
  const _Float16* r = base + (size_t)m * ld + koff;
  return combine16(*(const h8*)r, *(const h8*)(r + 8));
}

// B-operand (32x16) for x @ S^T with S f16 row-major [N][K]:
// lane n=lane&15, koff=16*(lane>>4); elements 0..15 = K koff+0..15.
__device__ __forceinline__ v16h frag_b_h(const _Float16* __restrict__ base,
                                         int ld, int lane) {
  const int n    = lane & 15;
  const int koff = (lane >> 4) << 4;
  const _Float16* r = base + (size_t)n * ld + koff;
  return combine16(*(const h8*)r, *(const h8*)(r + 8));
}

// ---------------------------------------------------------------------------
// Pack kernels: fp32 -> f16, run once per launch.
// ---------------------------------------------------------------------------
__global__ void __launch_bounds__(256) pack_f16_kernel(
    const float* __restrict__ src, _Float16* __restrict__ dst, int n4) {
  const int i = blockIdx.x * 256 + threadIdx.x;
  if (i < n4) {
    f4 f = ((const f4*)src)[i];
    h4 h;
#pragma unroll
    for (int j = 0; j < 4; ++j) h[j] = (_Float16)f[j];
    ((h4*)dst)[i] = h;
  }
}

// B[E,D,RANK] fp32 -> Bt[D][E*RANK] f16 with Bt[n][e*16+r] = B[e][n][r].
__global__ void __launch_bounds__(256) pack_bt_kernel(
    const float* __restrict__ Bm, _Float16* __restrict__ Bt) {
  const int i = blockIdx.x * 256 + threadIdx.x;  // i = e*1024 + n, over 8192
  const int e = i >> 10;
  const int n = i & 1023;
  const float* s = Bm + (size_t)i * RANK;
  _Float16* d = Bt + (size_t)n * ERDIM + e * RANK;
#pragma unroll
  for (int r = 0; r < RANK; r += 4) {
    f4 f = *(const f4*)(s + r);
    h4 h;
#pragma unroll
    for (int j = 0; j < 4; ++j) h[j] = (_Float16)f[j];
    *(h4*)(d + r) = h;
  }
}

// ---------------------------------------------------------------------------
// Kernel 1: router.  One wave per token; exact fp32 softmax + top-2 + renorm.
// ---------------------------------------------------------------------------
__global__ void __launch_bounds__(256) router_kernel(
    const float* __restrict__ x, const float* __restrict__ Rw,
    float* __restrict__ gate) {
  const int lane = threadIdx.x & 31;
  const int m    = blockIdx.x * 8 + (threadIdx.x >> 5);

  float acc[NEXP];
#pragma unroll
  for (int e = 0; e < NEXP; ++e) acc[e] = 0.f;

  const float* xr = x + (size_t)m * DDIM;
  for (int d = lane; d < DDIM; d += 32) {
    const float xv = xr[d];
#pragma unroll
    for (int e = 0; e < NEXP; ++e)
      acc[e] = fmaf(xv, Rw[e * DDIM + d], acc[e]);
  }
#pragma unroll
  for (int e = 0; e < NEXP; ++e)
    for (int off = 16; off > 0; off >>= 1)
      acc[e] += __shfl_xor(acc[e], off, 32);

  float mx = acc[0];
#pragma unroll
  for (int e = 1; e < NEXP; ++e) mx = fmaxf(mx, acc[e]);
  float p[NEXP], s = 0.f;
#pragma unroll
  for (int e = 0; e < NEXP; ++e) { p[e] = __expf(acc[e] - mx); s += p[e]; }
  float g[NEXP];
#pragma unroll
  for (int e = 0; e < NEXP; ++e) g[e] = p[e] / s;

  int i1 = 0; float v1 = g[0];
#pragma unroll
  for (int e = 1; e < NEXP; ++e) if (g[e] > v1) { v1 = g[e]; i1 = e; }
  int i2 = 0; float v2 = -1.f;
#pragma unroll
  for (int e = 0; e < NEXP; ++e)
    if (e != i1 && g[e] > v2) { v2 = g[e]; i2 = e; }
  const float t = v1 + v2 + 1e-6f;

  if (lane < NEXP)
    gate[(size_t)m * NEXP + lane] =
        (lane == i1 || lane == i2) ? g[lane] / t : 0.f;
}

// ---------------------------------------------------------------------------
// Kernel 2: gh = gate * (x @ A_all^T), output f16 [M,128].
// One wave per 16x16 tile; f16 operands.
// ---------------------------------------------------------------------------
__global__ void __launch_bounds__(256) lora_gh_kernel(
    const _Float16* __restrict__ xh, const _Float16* __restrict__ Ah,
    const float* __restrict__ gate, _Float16* __restrict__ gh) {
  const int lane  = threadIdx.x & 31;
  const int wg    = blockIdx.x * 8 + (threadIdx.x >> 5);
  const int mbase = (wg >> 3) << 4;
  const int nbase = (wg & 7) << 4;

  v8f c = {};
  for (int kb = 0; kb < DDIM; kb += 32) {
    v16h a  = frag_a_h(xh + (size_t)mbase * DDIM + kb, DDIM, lane);
    v16h bf = frag_b_h(Ah + (size_t)nbase * DDIM + kb, DDIM, lane);
    c = __builtin_amdgcn_wmma_f32_16x16x32_f16(false, a, false, bf,
                                               (short)0, c, false, false);
  }

  const int m0  = mbase + ((lane >> 4) << 3);
  const int col = nbase + (lane & 15);
  const int e   = col >> 4;
#pragma unroll
  for (int j = 0; j < 8; ++j) {
    const int m = m0 + j;
    gh[(size_t)m * ERDIM + col] = (_Float16)(c[j] * gate[(size_t)m * NEXP + e]);
  }
}

// ---------------------------------------------------------------------------
// Kernel 3: out = x @ W^T + b + gh @ Bt^T.
// Block = 8 waves = 128 rows x 64 cols; the 64x32 f16 W slice (4KB) is
// DMA'd into LDS with global_load_async_to_lds_b128, double-buffered:
//   wait_asynccnt(0) ; barrier ; async-fill nxt ; consume cur
// ---------------------------------------------------------------------------
__global__ void __launch_bounds__(256) fused_out_kernel(
    const _Float16* __restrict__ xh, const _Float16* __restrict__ Wh,
    const float* __restrict__ bvec, const _Float16* __restrict__ Bth,
    const _Float16* __restrict__ gh, float* __restrict__ out) {
  __shared__ _Float16 wtile[2][64 * 32];

  const int lane  = threadIdx.x & 31;
  const int wave  = threadIdx.x >> 5;
  const int nblk  = blockIdx.x & 15;          // 16 col groups of 64
  const int mblk  = blockIdx.x >> 4;          // 128 row groups of 128
  const int nbase = nblk << 6;
  const int mbase = (mblk << 7) + (wave << 4);

  // cooperative W stage: thread t -> row t>>2, 8 f16 at seg (t&3)*8
  const int srow = threadIdx.x >> 2;
  const int sseg = (threadIdx.x & 3) << 3;
  const _Float16* wsrc = Wh + (size_t)(nbase + srow) * DDIM + sseg;
  const unsigned ldst0 = lds_off32(&wtile[0][srow * 32 + sseg]);
  const unsigned ldst1 = lds_off32(&wtile[1][srow * 32 + sseg]);

  // prologue: async-fill buffer 0 with k=0
  async_fill_b128(ldst0, wsrc);

  v8f c0 = {}, c1 = {}, c2 = {}, c3 = {};
  const _Float16* xrow = xh + (size_t)mbase * DDIM;

  int cur = 0;
  for (int kb = 0; kb < DDIM; kb += 32) {
    wait_async0();        // my fill of `cur` has landed in LDS
    __syncthreads();      // all threads' fills landed; nxt's readers done
    if (kb + 32 < DDIM)
      async_fill_b128(cur ? ldst0 : ldst1, wsrc + kb + 32);

    v16h a = frag_a_h(xrow + kb, DDIM, lane);

    const int n    = lane & 15;
    const int koff = (lane >> 4) << 4;
#pragma unroll
    for (int sub = 0; sub < 4; ++sub) {
      const _Float16* lp = &wtile[cur][(sub * 16 + n) * 32 + koff];
      v16h bf = combine16(*(const h8*)lp, *(const h8*)(lp + 8));
      v8f* cc = (sub == 0) ? &c0 : (sub == 1) ? &c1 : (sub == 2) ? &c2 : &c3;
      *cc = __builtin_amdgcn_wmma_f32_16x16x32_f16(false, a, false, bf,
                                                   (short)0, *cc, false, false);
    }
    cur ^= 1;
  }

  // LoRA epilogue: K = 128 over gh (f16) and Bt (f16, [1024][128])
  for (int kb = 0; kb < ERDIM; kb += 32) {
    v16h a = frag_a_h(gh + (size_t)mbase * ERDIM + kb, ERDIM, lane);
    v16h b0 = frag_b_h(Bth + (size_t)(nbase +  0) * ERDIM + kb, ERDIM, lane);
    c0 = __builtin_amdgcn_wmma_f32_16x16x32_f16(false, a, false, b0,
                                                (short)0, c0, false, false);
    v16h b1 = frag_b_h(Bth + (size_t)(nbase + 16) * ERDIM + kb, ERDIM, lane);
    c1 = __builtin_amdgcn_wmma_f32_16x16x32_f16(false, a, false, b1,
                                                (short)0, c1, false, false);
    v16h b2 = frag_b_h(Bth + (size_t)(nbase + 32) * ERDIM + kb, ERDIM, lane);
    c2 = __builtin_amdgcn_wmma_f32_16x16x32_f16(false, a, false, b2,
                                                (short)0, c2, false, false);
    v16h b3 = frag_b_h(Bth + (size_t)(nbase + 48) * ERDIM + kb, ERDIM, lane);
    c3 = __builtin_amdgcn_wmma_f32_16x16x32_f16(false, a, false, b3,
                                                (short)0, c3, false, false);
  }

  v8f cc[4] = {c0, c1, c2, c3};
  const int rbase = (lane >> 4) << 3;
  const int n     = lane & 15;
#pragma unroll
  for (int sub = 0; sub < 4; ++sub) {
    const int col  = nbase + sub * 16 + n;
    const float bb = bvec[col];
#pragma unroll
    for (int j = 0; j < 8; ++j)
      out[(size_t)(mbase + rbase + j) * DDIM + col] = cc[sub][j] + bb;
  }
}

// ---------------------------------------------------------------------------
extern "C" void kernel_launch(void* const* d_in, const int* in_sizes, int n_in,
                              void* d_out, int out_size, void* d_ws,
                              size_t ws_size, hipStream_t stream) {
  (void)in_sizes; (void)n_in; (void)out_size; (void)ws_size;
  const float* x    = (const float*)d_in[0];  // [8,2048,1024]
  const float* W    = (const float*)d_in[1];  // [1024,1024]
  const float* bvec = (const float*)d_in[2];  // [1024]
  const float* Rw   = (const float*)d_in[3];  // [8,1024]
  const float* A    = (const float*)d_in[4];  // [8,16,1024] == A_all[128][1024]
  const float* Bm   = (const float*)d_in[5];  // [8,1024,16]
  float* out = (float*)d_out;                 // [8,2048,1024]

  // workspace carve-up (all chunks 256B-aligned multiples)
  char* ws = (char*)d_ws;
  float* gate = (float*)ws;            ws += (size_t)M_TOT * NEXP * 4;   // 512KB
  _Float16* gh  = (_Float16*)ws;       ws += (size_t)M_TOT * ERDIM * 2;  // 4MB
  _Float16* xh  = (_Float16*)ws;       ws += (size_t)M_TOT * DDIM * 2;   // 32MB
  _Float16* Wh  = (_Float16*)ws;       ws += (size_t)DDIM * DDIM * 2;    // 2MB
  _Float16* Ah  = (_Float16*)ws;       ws += (size_t)ERDIM * DDIM * 2;   // 256KB
  _Float16* Bth = (_Float16*)ws;       // 256KB

  // pack fp32 operands to f16 (once per launch)
  pack_f16_kernel<<<(M_TOT * DDIM / 4 + 255) / 256, 256, 0, stream>>>(
      x, xh, M_TOT * DDIM / 4);
  pack_f16_kernel<<<(DDIM * DDIM / 4 + 255) / 256, 256, 0, stream>>>(
      W, Wh, DDIM * DDIM / 4);
  pack_f16_kernel<<<(ERDIM * DDIM / 4 + 255) / 256, 256, 0, stream>>>(
      A, Ah, ERDIM * DDIM / 4);
  pack_bt_kernel<<<(NEXP * DDIM + 255) / 256, 256, 0, stream>>>(Bm, Bth);

  router_kernel<<<M_TOT / 8, 256, 0, stream>>>(x, Rw, gate);
  lora_gh_kernel<<<(M_TOT / 16) * (ERDIM / 16) / 8, 256, 0, stream>>>(
      xh, Ah, gate, gh);
  fused_out_kernel<<<(M_TOT / 128) * (DDIM / 64), 256, 0, stream>>>(
      xh, Wh, bvec, Bth, gh, out);
}
